// LocallyConnected2d_6373731467909
// MI455X (gfx1250) — matrix operand order repaired and checked by reading, not compiled, not run on hardware
//
#include <hip/hip_runtime.h>
#include <hip/hip_bf16.h>

typedef __attribute__((ext_vector_type(16))) __bf16 v16bf;
typedef __attribute__((ext_vector_type(8)))  __bf16 v8bf;
typedef __attribute__((ext_vector_type(8)))  float  v8f;
typedef __attribute__((ext_vector_type(4)))  float  v4f;

namespace {
constexpr int Bn = 32;    // batch
constexpr int Cc = 48;    // in channels
constexpr int On = 48;    // out channels
constexpr int Kk = 432;   // C*kh*kw
constexpr int KP = 448;   // K padded to multiple of 32
constexpr int Ll = 4096;  // Hout*Wout
constexpr int THREADS = 96;  // 3 waves: one o-tile each, both b-tiles
}

__global__ __launch_bounds__(THREADS) void lc2d_wmma_bf16(
    const float* __restrict__ x,       // [32,48,64,64]
    const float* __restrict__ weight,  // [48,64,64,48,3,3] == [O, L, K]
    const float* __restrict__ bias,    // [48]
    float* __restrict__ out)           // [32,48,64,64]
{
  __shared__ __align__(16) __bf16 Alds[Bn * KP];  // 28 KB

  const int l   = blockIdx.x;          // spatial location
  const int ho  = l >> 6;
  const int wo  = l & 63;
  const int tid = threadIdx.x;

  // ---- Stage A = unfolded x patch for this location: [32 x 448] bf16 (k>=432 -> 0).
  // x is fully L2-resident (25 MB < 192 MB) and reused 9x spatially: keep RT hints.
  for (int idx = tid; idx < Bn * KP; idx += THREADS) {
    const int b = idx / KP;
    const int k = idx - b * KP;
    float v = 0.f;
    if (k < Kk) {
      const int c  = k / 9;
      const int r9 = k - c * 9;
      const int r  = r9 / 3;
      const int s  = r9 - r * 3;
      const int h  = ho - 1 + r;
      const int w  = wo - 1 + s;
      if ((unsigned)h < 64u && (unsigned)w < 64u)
        v = x[(((size_t)b * Cc + c) << 12) + (h << 6) + w];
    }
    Alds[idx] = (__bf16)v;
  }
  __syncthreads();

  const int lane = tid & 31;
  const int nt   = tid >> 5;       // o-tile 0..2 (one per wave)
  const int lo16 = lane & 15;
  const int hi   = lane >> 4;      // k-half selector within a 32-wide step

  const int arow0 = lo16;          // b-tile 0 rows
  const int arow1 = 16 + lo16;     // b-tile 1 rows
  const int brow  = nt * 16 + lo16;
  const float* __restrict__ wrow = weight + ((size_t)brow * Ll + l) * Kk;

  v8f acc0 = {};
  v8f acc1 = {};

#pragma unroll
  for (int s = 0; s < KP / 32; ++s) {
    // A fragments (this code's k order, same as B; not AMD's table order): lane holds K = hi*16..+15 (contiguous)
    const int ka = s * 32 + (hi << 4);
    v8bf a00 = *(const v8bf*)(&Alds[arow0 * KP + ka]);
    v8bf a01 = *(const v8bf*)(&Alds[arow0 * KP + ka + 8]);
    v8bf a10 = *(const v8bf*)(&Alds[arow1 * KP + ka]);
    v8bf a11 = *(const v8bf*)(&Alds[arow1 * KP + ka + 8]);
    v16bf av0 = __builtin_shufflevector(a00, a01, 0,1,2,3,4,5,6,7,8,9,10,11,12,13,14,15);
    v16bf av1 = __builtin_shufflevector(a10, a11, 0,1,2,3,4,5,6,7,8,9,10,11,12,13,14,15);

    // B fragment streamed straight from HBM, used exactly once device-wide:
    // non-temporal loads keep the 340 MB weight stream from evicting x out of L2.
    v16bf bv = {};
    if (s < (Kk / 32) || hi == 0) {     // last step's hi=1 half is the zero K-pad
      const float* wp = wrow + s * 32 + (hi << 4);
      const v4f w0 = __builtin_nontemporal_load((const v4f*)(wp + 0));
      const v4f w1 = __builtin_nontemporal_load((const v4f*)(wp + 4));
      const v4f w2 = __builtin_nontemporal_load((const v4f*)(wp + 8));
      const v4f w3 = __builtin_nontemporal_load((const v4f*)(wp + 12));
      bv[0]  = (__bf16)w0[0]; bv[1]  = (__bf16)w0[1]; bv[2]  = (__bf16)w0[2]; bv[3]  = (__bf16)w0[3];
      bv[4]  = (__bf16)w1[0]; bv[5]  = (__bf16)w1[1]; bv[6]  = (__bf16)w1[2]; bv[7]  = (__bf16)w1[3];
      bv[8]  = (__bf16)w2[0]; bv[9]  = (__bf16)w2[1]; bv[10] = (__bf16)w2[2]; bv[11] = (__bf16)w2[3];
      bv[12] = (__bf16)w3[0]; bv[13] = (__bf16)w3[1]; bv[14] = (__bf16)w3[2]; bv[15] = (__bf16)w3[3];
    }

    acc0 = __builtin_amdgcn_wmma_f32_16x16x32_bf16(
        false, av0, false, bv, (short)0, acc0, false, false);
    acc1 = __builtin_amdgcn_wmma_f32_16x16x32_bf16(
        false, av1, false, bv, (short)0, acc1, false, false);
  }

  // ---- Epilogue: C/D layout — VGPR v: lanes0-15 M=v, lanes16-31 M=v+8; N=lane%16
  // Output is written once and never re-read: non-temporal stores.
  const int o  = nt * 16 + lo16;
  const float bo = bias[o];
#pragma unroll
  for (int v = 0; v < 8; ++v) {
    const int bi = v + hi * 8;
    __builtin_nontemporal_store(acc0[v] + bo,
        &out[(((size_t)bi * On) + o) * Ll + l]);
    __builtin_nontemporal_store(acc1[v] + bo,
        &out[((((size_t)bi + 16) * On) + o) * Ll + l]);
  }
}

extern "C" void kernel_launch(void* const* d_in, const int* in_sizes, int n_in,
                              void* d_out, int out_size, void* d_ws, size_t ws_size,
                              hipStream_t stream) {
  (void)in_sizes; (void)n_in; (void)d_ws; (void)ws_size; (void)out_size;
  const float* x      = (const float*)d_in[0];
  const float* weight = (const float*)d_in[1];
  const float* bias   = (const float*)d_in[2];
  float* out          = (float*)d_out;

  lc2d_wmma_bf16<<<dim3(Ll), dim3(THREADS), 0, stream>>>(x, weight, bias, out);
}
